// TankBindTrigonometry_90142773608735
// MI455X (gfx1250) — compile-verified, matrix operand order checked by reading, not run on hardware
//
#include <hip/hip_runtime.h>
#include <hip/hip_bf16.h>

typedef __attribute__((ext_vector_type(16))) _Float16 v16h;
typedef __attribute__((ext_vector_type(8)))  _Float16 v8h;
typedef __attribute__((ext_vector_type(4)))  _Float16 v4h;
typedef __attribute__((ext_vector_type(8)))  float    v8f;

#define NPD   512
#define NCD   128
#define EDIM  128
#define HH    4
#define FFD   512
#define MZ    (NPD*NCD)     /* 65536  z rows       */
#define MPP   (NPD*NPD)     /* 262144 pair rows    */
#define MCP   (NCD*NCD)     /* 16384  compound rows*/

#if defined(__has_builtin)
#  if __has_builtin(__builtin_amdgcn_global_load_async_to_lds_b128)
#    define HAVE_ASYNC_LDS 1
#  else
#    define HAVE_ASYNC_LDS 0
#  endif
#  if __has_builtin(__builtin_amdgcn_s_wait_asynccnt)
#    define HAVE_WAIT_ASYNC 1
#  else
#    define HAVE_WAIT_ASYNC 0
#  endif
#else
#  define HAVE_ASYNC_LDS 0
#  define HAVE_WAIT_ASYNC 0
#endif

#if HAVE_ASYNC_LDS
typedef __attribute__((address_space(1))) void as1_void;
typedef __attribute__((address_space(3))) void as3_void;
typedef int v4i_g __attribute__((vector_size(16)));   // matches builtin's pointee type
typedef __attribute__((address_space(1))) v4i_g as1_v4i;
typedef __attribute__((address_space(3))) v4i_g as3_v4i;
#endif

// ---------------------------------------------------------------------------
// WMMA fragment helpers (gfx1250 wave32, 16x16x32 f16 -> f32)
// A-fragment / Bt-fragment from row-major [row][K] storage:
//   lane L(0..15)  : row L,    K = k0+0..7   and k0+16..23
//   lane L(16..31) : row L-16, K = k0+8..15  and k0+24..31
// ---------------------------------------------------------------------------
__device__ __forceinline__ v16h frag_row(const _Float16* __restrict__ p, int ld, int k0) {
  int lane = threadIdx.x & 31;
  int row  = lane & 15;
  int kh   = (lane >> 4) << 3;
  const _Float16* q = p + (size_t)row * ld + k0 + kh;
  v8h lo = *(const v8h*)q;
  v8h hi = *(const v8h*)(q + 16);
  v16h f;
#pragma unroll
  for (int t = 0; t < 8; ++t) { f[t] = lo[t]; f[t + 8] = hi[t]; }
  return f;
}

// Bt-fragment gathered from row-major [K][N] storage (implicit transpose)
__device__ __forceinline__ v16h frag_col(const _Float16* __restrict__ p, int ld, int k0) {
  int lane = threadIdx.x & 31;
  int n    = lane & 15;
  int kh   = (lane >> 4) << 3;
  v16h f;
#pragma unroll
  for (int t = 0; t < 8; ++t) {
    f[t]     = p[(size_t)(k0 + kh + t)      * ld + n];
    f[t + 8] = p[(size_t)(k0 + kh + 16 + t) * ld + n];
  }
  return f;
}

__device__ __forceinline__ v8f wmma_f16(v16h a, v16h b, v8f c) {
  return __builtin_amdgcn_wmma_f32_16x16x32_f16(false, a, false, b, (short)0, c,
                                                false, false);
}

__device__ __forceinline__ float sigmoidf_(float x) { return 1.0f / (1.0f + __expf(-x)); }

// ---------------------------------------------------------------------------
// Weight transpose+convert: in f32 [K][N] row-major -> out f16 [N][K]
// ---------------------------------------------------------------------------
__global__ void wtrans_kernel(const float* __restrict__ in, _Float16* __restrict__ out,
                              int K, int N) {
  int idx = blockIdx.x * blockDim.x + threadIdx.x;
  if (idx >= K * N) return;
  int n = idx / K, k = idx % K;
  out[idx] = (_Float16)in[(size_t)k * N + n];
}

// ---------------------------------------------------------------------------
// LayerNorm over last dim (128) -> f16. One wave per row, shfl reduction.
// ---------------------------------------------------------------------------
__global__ void ln_f16_kernel(const float* __restrict__ x, const float* __restrict__ w,
                              const float* __restrict__ b, _Float16* __restrict__ y,
                              int Mrows) {
  int gid  = blockIdx.x * blockDim.x + threadIdx.x;
  int row  = gid >> 5;
  int lane = threadIdx.x & 31;
  if (row >= Mrows) return;
  const float4 xv = *(const float4*)(x + (size_t)row * 128 + lane * 4);
  float s  = xv.x + xv.y + xv.z + xv.w;
  float s2 = xv.x * xv.x + xv.y * xv.y + xv.z * xv.z + xv.w * xv.w;
#pragma unroll
  for (int o = 16; o >= 1; o >>= 1) {
    s  += __shfl_xor(s,  o, 32);
    s2 += __shfl_xor(s2, o, 32);
  }
  float mean = s * (1.0f / 128.0f);
  float var  = s2 * (1.0f / 128.0f) - mean * mean;
  float r    = rsqrtf(var + 1e-5f);
  v4h out;
#pragma unroll
  for (int t = 0; t < 4; ++t) {
    int   c  = lane * 4 + t;
    float xi = (t == 0 ? xv.x : t == 1 ? xv.y : t == 2 ? xv.z : xv.w);
    float val = (xi - mean) * r * w[c] + (b ? b[c] : 0.0f);
    out[t] = (_Float16)val;
  }
  *(v4h*)(y + (size_t)row * 128 + lane * 4) = out;
}

// ---------------------------------------------------------------------------
// Dual GEMM: out = sigmoid(A@G) * (A@L + lbias) [* mask]  -> f16
// 2x2 register blocking: each wave computes a 32x32 tile (8 WMMAs / k-step).
// OUTMODE 1: channel-major  out[c*M + m]
// OUTMODE 2: channel-major + per-plane transpose (512x128 -> 128x512)
// ---------------------------------------------------------------------------
template <int OUTMODE>
__global__ void gatepair_kernel(const _Float16* __restrict__ A, const _Float16* __restrict__ Gt,
                                const _Float16* __restrict__ Lt, const float* __restrict__ lbias,
                                const float* __restrict__ mask, _Float16* __restrict__ out,
                                int M, int K) {
  int mt = blockIdx.x, nt = blockIdx.y;
  const _Float16* Ap = A  + (size_t)mt * 32 * K;
  const _Float16* Gp = Gt + (size_t)nt * 32 * K;
  const _Float16* Lp = Lt + (size_t)nt * 32 * K;
  v8f ag[2][2] = {};
  v8f al[2][2] = {};
  for (int kk = 0; kk < K; kk += 32) {
    v16h a0 = frag_row(Ap, K, kk);
    v16h a1 = frag_row(Ap + (size_t)16 * K, K, kk);
#pragma unroll
    for (int n = 0; n < 2; ++n) {
      v16h bg = frag_row(Gp + (size_t)n * 16 * K, K, kk);
      ag[0][n] = wmma_f16(a0, bg, ag[0][n]);
      ag[1][n] = wmma_f16(a1, bg, ag[1][n]);
      v16h bl = frag_row(Lp + (size_t)n * 16 * K, K, kk);
      al[0][n] = wmma_f16(a0, bl, al[0][n]);
      al[1][n] = wmma_f16(a1, bl, al[1][n]);
    }
  }
  int lane = threadIdx.x & 31;
  int hi8  = (lane >> 4) << 3;
  int c15  = lane & 15;
#pragma unroll
  for (int m2 = 0; m2 < 2; ++m2) {
#pragma unroll
    for (int n2 = 0; n2 < 2; ++n2) {
#pragma unroll
      for (int vv = 0; vv < 8; ++vv) {
        int row = mt * 32 + m2 * 16 + vv + hi8;
        int col = nt * 32 + n2 * 16 + c15;
        float val = sigmoidf_(ag[m2][n2][vv]) * (al[m2][n2][vv] + lbias[col]);
        if (mask) val *= mask[row];
        size_t idx;
        if (OUTMODE == 1) idx = (size_t)col * M + row;
        else              idx = (size_t)col * M + (size_t)(row & 127) * 512 + (row >> 7);
        out[idx] = (_Float16)val;
      }
    }
  }
}

// ---------------------------------------------------------------------------
// Generic GEMM  C = A[M,K] @ Bt[N,K]^T  with 2x4 register blocking
// (32x64 tile per wave, 8 WMMAs per 12 b128 loads). Epilogues:
//   EPI 0: val = acc + bias
//   EPI 1: val = sigmoid(acc + bias)
//   EPI 2: val = resid + gmul * (acc + bias) * mask[row]     (tri update)
//   EPI 3: val = resid + (acc + bias) * mask[row]            (attn residual)
//   EPI 4: val = relu(acc + bias)                            (final output)
// ---------------------------------------------------------------------------
template <int EPI, bool F16OUT>
__global__ void gemm16_kernel(const _Float16* __restrict__ A, const _Float16* __restrict__ Bt,
                              const float* __restrict__ bias, const float* __restrict__ gmul,
                              const float* __restrict__ resid, const float* __restrict__ mask,
                              void* __restrict__ outp, int N, int K) {
  int mt = blockIdx.x, nt = blockIdx.y;
  const _Float16* Ap = A  + (size_t)mt * 32 * K;
  const _Float16* Bp = Bt + (size_t)nt * 64 * K;
  v8f acc[2][4] = {};
  for (int kk = 0; kk < K; kk += 32) {
    v16h a0 = frag_row(Ap, K, kk);
    v16h a1 = frag_row(Ap + (size_t)16 * K, K, kk);
#pragma unroll
    for (int n = 0; n < 4; ++n) {
      v16h b = frag_row(Bp + (size_t)n * 16 * K, K, kk);
      acc[0][n] = wmma_f16(a0, b, acc[0][n]);
      acc[1][n] = wmma_f16(a1, b, acc[1][n]);
    }
  }
  int lane = threadIdx.x & 31;
  int hi8  = (lane >> 4) << 3;
  int c15  = lane & 15;
#pragma unroll
  for (int m2 = 0; m2 < 2; ++m2) {
#pragma unroll
    for (int n2 = 0; n2 < 4; ++n2) {
#pragma unroll
      for (int vv = 0; vv < 8; ++vv) {
        int row = mt * 32 + m2 * 16 + vv + hi8;
        int col = nt * 64 + n2 * 16 + c15;
        size_t idx = (size_t)row * N + col;
        float val = acc[m2][n2][vv] + (bias ? bias[col] : 0.0f);
        if (EPI == 1) val = sigmoidf_(val);
        if (EPI == 2) val = resid[idx] + gmul[idx] * val * mask[row];
        if (EPI == 3) val = resid[idx] + val * mask[row];
        if (EPI == 4) val = fmaxf(val, 0.0f);
        if (F16OUT) ((_Float16*)outp)[idx] = (_Float16)val;
        else        ((float*)outp)[idx]    = val;
      }
    }
  }
}

// ---------------------------------------------------------------------------
// Triangle einsum, per channel c:
//   block[:,:,c] = PP_c[512,512] @ AB1_c[512,128] + AB2_c[512,128] @ CP_c^T
// Channel-major f16 planes: ppt [c][i][k] ld512, ab1t [c][j][k] ld512,
// ab2t [c][i][k] ld128, cpt [c][j][k] ld128.
// One wave per (c, 32-row i-block, 64-col j-block): 160 WMMAs per wave.
// ---------------------------------------------------------------------------
__global__ void block_einsum_kernel(const _Float16* __restrict__ ppt,
                                    const _Float16* __restrict__ ab1t,
                                    const _Float16* __restrict__ ab2t,
                                    const _Float16* __restrict__ cpt,
                                    float* __restrict__ outb) {
  int jt = blockIdx.x;   // 0..1   (64-col block)
  int it = blockIdx.y;   // 0..15  (32-row block)
  int c  = blockIdx.z;   // 0..127
  v8f acc[2][4] = {};
  const _Float16* Ap = ppt  + (size_t)c * ((size_t)NPD * NPD) + (size_t)it * 32 * 512;
  const _Float16* Bp = ab1t + (size_t)c * ((size_t)NCD * NPD) + (size_t)jt * 64 * 512;
  for (int kk = 0; kk < 512; kk += 32) {
    v16h a0 = frag_row(Ap, 512, kk);
    v16h a1 = frag_row(Ap + (size_t)16 * 512, 512, kk);
#pragma unroll
    for (int n = 0; n < 4; ++n) {
      v16h b = frag_row(Bp + (size_t)n * 16 * 512, 512, kk);
      acc[0][n] = wmma_f16(a0, b, acc[0][n]);
      acc[1][n] = wmma_f16(a1, b, acc[1][n]);
    }
  }
  const _Float16* A2 = ab2t + (size_t)c * ((size_t)NPD * NCD) + (size_t)it * 32 * 128;
  const _Float16* B2 = cpt  + (size_t)c * ((size_t)NCD * NCD) + (size_t)jt * 64 * 128;
  for (int kk = 0; kk < 128; kk += 32) {
    v16h a0 = frag_row(A2, 128, kk);
    v16h a1 = frag_row(A2 + (size_t)16 * 128, 128, kk);
#pragma unroll
    for (int n = 0; n < 4; ++n) {
      v16h b = frag_row(B2 + (size_t)n * 16 * 128, 128, kk);
      acc[0][n] = wmma_f16(a0, b, acc[0][n]);
      acc[1][n] = wmma_f16(a1, b, acc[1][n]);
    }
  }
  int lane = threadIdx.x & 31;
  int hi8  = (lane >> 4) << 3;
  int c15  = lane & 15;
#pragma unroll
  for (int m2 = 0; m2 < 2; ++m2) {
#pragma unroll
    for (int n2 = 0; n2 < 4; ++n2) {
#pragma unroll
      for (int vv = 0; vv < 8; ++vv) {
        int i = it * 32 + m2 * 16 + vv + hi8;
        int j = jt * 64 + n2 * 16 + c15;
        outb[((size_t)i * 128 + j) * 128 + c] = acc[m2][n2][vv];
      }
    }
  }
}

// ---------------------------------------------------------------------------
// Self-attention per (protein i, head h). 4 waves, each owns 2 row-tiles.
// V head-block (128x32 f16) is staged to LDS via async global->LDS copies
// (ASYNCcnt path) when the toolchain exposes the builtin; S@V B-fragments
// then gather from LDS instead of strided HBM.
// ---------------------------------------------------------------------------
__global__ void attention_kernel(const _Float16* __restrict__ q, const _Float16* __restrict__ k,
                                 const _Float16* __restrict__ v, const float* __restrict__ gate,
                                 const float* __restrict__ maskatt, _Float16* __restrict__ gwa) {
  int i    = blockIdx.x;
  int h    = blockIdx.y;
  int w    = threadIdx.x >> 5;
  int lane = threadIdx.x & 31;
  int hi8  = (lane >> 4) << 3;
  int c15  = lane & 15;
  __shared__ __align__(16) _Float16 sP[4][16 * 128];
  __shared__ __align__(16) _Float16 sV[128 * 32];
  const float scale = 0.17677669529663687f;  // 1/sqrt(32)

  // ---- stage V[i, :, h*32 : h*32+32] into LDS ----
  {
    const _Float16* vhead = v + ((size_t)i * 128) * 128 + h * 32;
    int t = threadIdx.x;
#pragma unroll
    for (int j = 0; j < 4; ++j) {
      int idx = j * 128 + t;       // 0..511 : 16-byte chunks
      int kr  = idx >> 2;          // V row (0..127)
      int ch  = idx & 3;           // chunk within row
      const _Float16* g = vhead + (size_t)kr * 128 + ch * 8;
      _Float16*       l = &sV[kr * 32 + ch * 8];
#if HAVE_ASYNC_LDS
      __builtin_amdgcn_global_load_async_to_lds_b128(
          (as1_v4i*)(as1_void*)g, (as3_v4i*)(as3_void*)l, 0, 0);
#else
      *(v8h*)l = *(const v8h*)g;
#endif
    }
#if HAVE_ASYNC_LDS && HAVE_WAIT_ASYNC
    __builtin_amdgcn_s_wait_asynccnt(0);
#endif
    __syncthreads();
  }

  for (int rr = 0; rr < 2; ++rr) {
    int rt = w * 2 + rr;
    v16h aq = frag_row(q + ((size_t)(i * 128 + rt * 16)) * 128 + h * 32, 128, 0);
    v8f acc[8];
#pragma unroll
    for (int ct = 0; ct < 8; ++ct) {
      v16h bk = frag_row(k + ((size_t)(i * 128 + ct * 16)) * 128 + h * 32, 128, 0);
      v8f zro = {};
      acc[ct] = wmma_f16(aq, bk, zro);
    }
#pragma unroll
    for (int vv = 0; vv < 8; ++vv) {
      int qrow = rt * 16 + vv + hi8;
      size_t mbase = (((size_t)i * HH + h) * 128 + qrow) * 128;
      float mx = -3.0e38f;
#pragma unroll
      for (int ct = 0; ct < 8; ++ct) {
        float l = acc[ct][vv] * scale + maskatt[mbase + ct * 16 + c15];
        acc[ct][vv] = l;
        mx = fmaxf(mx, l);
      }
#pragma unroll
      for (int o = 1; o < 16; o <<= 1) mx = fmaxf(mx, __shfl_xor(mx, o, 16));
      float s = 0.0f;
#pragma unroll
      for (int ct = 0; ct < 8; ++ct) {
        float e = __expf(acc[ct][vv] - mx);
        acc[ct][vv] = e;
        s += e;
      }
#pragma unroll
      for (int o = 1; o < 16; o <<= 1) s += __shfl_xor(s, o, 16);
      float inv = 1.0f / s;
#pragma unroll
      for (int ct = 0; ct < 8; ++ct)
        sP[w][(vv + hi8) * 128 + ct * 16 + c15] = (_Float16)(acc[ct][vv] * inv);
    }
    __syncthreads();

    // S @ V : A fragment reused across both 16-col N-tiles
    v8f acco[2] = {};
    for (int kk = 0; kk < 128; kk += 32) {
      v16h as = frag_row(&sP[w][0], 128, kk);
#pragma unroll
      for (int nt2 = 0; nt2 < 2; ++nt2) {
        v16h bv = frag_col(&sV[nt2 * 16], 32, kk);
        acco[nt2] = wmma_f16(as, bv, acco[nt2]);
      }
    }
#pragma unroll
    for (int nt2 = 0; nt2 < 2; ++nt2) {
#pragma unroll
      for (int vv = 0; vv < 8; ++vv) {
        size_t m  = (size_t)i * 128 + rt * 16 + vv + hi8;
        int   col = h * 32 + nt2 * 16 + c15;
        float gv  = gate[m * 128 + col];
        gwa[m * 128 + col] = (_Float16)(gv * acco[nt2][vv]);
      }
    }
    __syncthreads();
  }
}

// ---------------------------------------------------------------------------
// Host driver
// ---------------------------------------------------------------------------
extern "C" void kernel_launch(void* const* d_in, const int* in_sizes, int n_in,
                              void* d_out, int out_size, void* d_ws, size_t ws_size,
                              hipStream_t stream) {
  (void)in_sizes; (void)n_in; (void)out_size; (void)ws_size;

  const float* z_in      = (const float*)d_in[0];
  const float* pp_in     = (const float*)d_in[1];
  const float* cp_in     = (const float*)d_in[2];
  const float* z_mask    = (const float*)d_in[3];
  const float* zmatt     = (const float*)d_in[4];
  const float* ptc_ln_w  = (const float*)d_in[5];
  const float* ptc_lnc_w = (const float*)d_in[6];
  const float* ptc_g1_w  = (const float*)d_in[7];
  const float* ptc_g2_w  = (const float*)d_in[8];
  const float* ptc_l1_w  = (const float*)d_in[9];
  const float* ptc_l1_b  = (const float*)d_in[10];
  const float* ptc_l2_w  = (const float*)d_in[11];
  const float* ptc_l2_b  = (const float*)d_in[12];
  const float* ptc_eg_w  = (const float*)d_in[13];
  const float* ptc_eg_b  = (const float*)d_in[14];
  const float* ptc_as_w  = (const float*)d_in[15];
  const float* ptc_as_b  = (const float*)d_in[16];
  const float* att_ln_w  = (const float*)d_in[17];
  const float* att_ln_b  = (const float*)d_in[18];
  const float* att_q_w   = (const float*)d_in[19];
  const float* att_k_w   = (const float*)d_in[20];
  const float* att_v_w   = (const float*)d_in[21];
  const float* att_g_w   = (const float*)d_in[22];
  const float* att_g_b   = (const float*)d_in[23];
  const float* att_o_w   = (const float*)d_in[24];
  const float* att_o_b   = (const float*)d_in[25];
  const float* tr_ln_w   = (const float*)d_in[26];
  const float* tr_ln_b   = (const float*)d_in[27];
  const float* tr_l1_w   = (const float*)d_in[28];
  const float* tr_l1_b   = (const float*)d_in[29];
  const float* tr_l2_w   = (const float*)d_in[30];
  const float* tr_l2_b   = (const float*)d_in[31];

  // ---- workspace layout (lifetime-based reuse, ~345 MB peak) ----
  char*  W   = (char*)d_ws;
  size_t off = 0;
  auto take = [&](size_t bytes) -> char* {
    char* p = W + off;
    off += (bytes + 255) & ~(size_t)255;
    return p;
  };
  const size_t HB = sizeof(_Float16);

  _Float16* g1t = (_Float16*)take(16384 * HB);
  _Float16* g2t = (_Float16*)take(16384 * HB);
  _Float16* l1t = (_Float16*)take(16384 * HB);
  _Float16* l2t = (_Float16*)take(16384 * HB);
  _Float16* egt = (_Float16*)take(16384 * HB);
  _Float16* ast = (_Float16*)take(16384 * HB);
  _Float16* qt  = (_Float16*)take(16384 * HB);
  _Float16* kt  = (_Float16*)take(16384 * HB);
  _Float16* vt  = (_Float16*)take(16384 * HB);
  _Float16* gtt = (_Float16*)take(16384 * HB);
  _Float16* ot  = (_Float16*)take(16384 * HB);
  _Float16* tr1t = (_Float16*)take(65536 * HB);   // [512][128]
  _Float16* tr2t = (_Float16*)take(65536 * HB);   // [128][512]

  _Float16* ppn  = (_Float16*)take((size_t)MPP * 128 * HB);  // 64MB; reused as h1
  _Float16* ppt  = (_Float16*)take((size_t)MPP * 128 * HB);  // 64MB
  _Float16* znX  = (_Float16*)take((size_t)MZ * 128 * HB);   // 16MB; zn -> zn2 -> zn3
  _Float16* ab1t = (_Float16*)take((size_t)MZ * 128 * HB);   // 16MB
  _Float16* ab2t = (_Float16*)take((size_t)MZ * 128 * HB);   // 16MB
  _Float16* cpn  = (_Float16*)take((size_t)MCP * 128 * HB);  // 4MB
  _Float16* cpt  = (_Float16*)take((size_t)MCP * 128 * HB);  // 4MB
  float*    gbuf = (float*)take((size_t)MZ * 128 * 4);       // 32MB; reused as gate
  float*    blkb = (float*)take((size_t)MZ * 128 * 4);       // 32MB; reused as z2
  _Float16* blkn = (_Float16*)take((size_t)MZ * 128 * HB);   // 16MB; reused as qb
  float*    z1   = (float*)take((size_t)MZ * 128 * 4);       // 32MB
  _Float16* kb   = (_Float16*)take((size_t)MZ * 128 * HB);   // 16MB
  _Float16* vb   = (_Float16*)take((size_t)MZ * 128 * HB);   // 16MB
  _Float16* gwa  = (_Float16*)take((size_t)MZ * 128 * HB);   // 16MB

  _Float16* h1    = ppn;   // reuse 64MB region after ppt consumed
  float*    gateb = gbuf;  // reuse after tri epilogue
  float*    z2    = blkb;  // reuse after block LN
  _Float16* qb    = blkn;  // reuse after tri epilogue

  auto wt = [&](const float* src, _Float16* dst, int K2, int N2) {
    int tot = K2 * N2;
    wtrans_kernel<<<dim3((tot + 255) / 256), dim3(256), 0, stream>>>(src, dst, K2, N2);
  };
  auto ln = [&](const float* x, const float* w2, const float* b2, _Float16* y, int Mrows) {
    ln_f16_kernel<<<dim3(Mrows / 8), dim3(256), 0, stream>>>(x, w2, b2, y, Mrows);
  };
  dim3 wv(32);

  // 1) weights -> f16 transposed
  wt(ptc_g1_w, g1t, 128, 128);  wt(ptc_g2_w, g2t, 128, 128);
  wt(ptc_l1_w, l1t, 128, 128);  wt(ptc_l2_w, l2t, 128, 128);
  wt(ptc_eg_w, egt, 128, 128);  wt(ptc_as_w, ast, 128, 128);
  wt(att_q_w, qt, 128, 128);    wt(att_k_w, kt, 128, 128);
  wt(att_v_w, vt, 128, 128);    wt(att_g_w, gtt, 128, 128);
  wt(att_o_w, ot, 128, 128);
  wt(tr_l1_w, tr1t, 128, 512);  wt(tr_l2_w, tr2t, 512, 128);

  // 2) LayerNorms (ptc_ln, no bias)
  ln(z_in,  ptc_ln_w, nullptr, znX, MZ);
  ln(pp_in, ptc_ln_w, nullptr, ppn, MPP);
  ln(cp_in, ptc_ln_w, nullptr, cpn, MCP);

  // 3) gated projections (channel-major outputs for the triangle einsum)
  gatepair_kernel<1><<<dim3(MPP / 32, 4), wv, 0, stream>>>(ppn, g2t, l2t, ptc_l2_b, nullptr, ppt,  MPP, 128);
  gatepair_kernel<1><<<dim3(MCP / 32, 4), wv, 0, stream>>>(cpn, g1t, l1t, ptc_l1_b, nullptr, cpt,  MCP, 128);
  gatepair_kernel<2><<<dim3(MZ  / 32, 4), wv, 0, stream>>>(znX, g1t, l1t, ptc_l1_b, z_mask, ab1t, MZ, 128);
  gatepair_kernel<1><<<dim3(MZ  / 32, 4), wv, 0, stream>>>(znX, g2t, l2t, ptc_l2_b, z_mask, ab2t, MZ, 128);
  gemm16_kernel<1, false><<<dim3(MZ / 32, 2), wv, 0, stream>>>(znX, egt, ptc_eg_b, nullptr, nullptr,
                                                               nullptr, gbuf, 128, 128);

  // 4) triangle einsum (per-channel batched GEMMs, 32x64 tiles)
  block_einsum_kernel<<<dim3(2, 16, 128), wv, 0, stream>>>(ppt, ab1t, ab2t, cpt, blkb);

  // 5) tri epilogue: z1 = z + g * (LN_c(block) @ as + b) * mask
  ln(blkb, ptc_lnc_w, nullptr, blkn, MZ);
  gemm16_kernel<2, false><<<dim3(MZ / 32, 2), wv, 0, stream>>>(blkn, ast, ptc_as_b, gbuf, z_in,
                                                               z_mask, z1, 128, 128);

  // 6) attention: LN, Q/K/V/gate projections, core, output projection
  ln(z1, att_ln_w, att_ln_b, znX, MZ);
  gemm16_kernel<0, true><<<dim3(MZ / 32, 2), wv, 0, stream>>>(znX, qt, nullptr, nullptr, nullptr,
                                                              nullptr, qb, 128, 128);
  gemm16_kernel<0, true><<<dim3(MZ / 32, 2), wv, 0, stream>>>(znX, kt, nullptr, nullptr, nullptr,
                                                              nullptr, kb, 128, 128);
  gemm16_kernel<0, true><<<dim3(MZ / 32, 2), wv, 0, stream>>>(znX, vt, nullptr, nullptr, nullptr,
                                                              nullptr, vb, 128, 128);
  gemm16_kernel<1, false><<<dim3(MZ / 32, 2), wv, 0, stream>>>(znX, gtt, att_g_b, nullptr, nullptr,
                                                               nullptr, gateb, 128, 128);
  attention_kernel<<<dim3(NPD, HH), dim3(128), 0, stream>>>(qb, kb, vb, gateb, zmatt, gwa);
  gemm16_kernel<3, false><<<dim3(MZ / 32, 2), wv, 0, stream>>>(gwa, ot, att_o_b, nullptr, z1,
                                                               z_mask, z2, 128, 128);

  // 7) transition: relu((LN(z2) @ l1 + b1) @ l2 + b2) -> d_out
  ln(z2, tr_ln_w, tr_ln_b, znX, MZ);
  gemm16_kernel<0, true><<<dim3(MZ / 32, FFD / 64), wv, 0, stream>>>(znX, tr1t, tr_l1_b, nullptr,
                                                                     nullptr, nullptr, h1, FFD, 128);
  gemm16_kernel<4, false><<<dim3(MZ / 32, 2), wv, 0, stream>>>(h1, tr2t, tr_l2_b, nullptr, nullptr,
                                                               nullptr, (float*)d_out, 128, FFD);
}
